// CustomWavLMEncoderLayer_1125281431981
// MI455X (gfx1250) — compile-verified, matrix operand order checked by reading, not run on hardware
//
#include <hip/hip_runtime.h>
#include <hip/hip_bf16.h>
#include <math.h>

#define H_  1024
#define NH_ 16
#define HD_ 64
#define FF_ 4096
#define AD_ 256
#define NB_ 320
#define MAXD_ 800
#define B_  4
#define S_  1024
#define EPS_ 1e-5f

typedef __attribute__((ext_vector_type(16))) _Float16 v16h;
typedef __attribute__((ext_vector_type(8)))  float    v8f;

static __device__ __forceinline__ v8f wmma_f16(v16h a, v16h b, v8f c) {
  // D = A(16x32 f16) * B(32x16 f16) + C(16x16 f32)
  return __builtin_amdgcn_wmma_f32_16x16x32_f16(false, a, false, b, (short)0, c,
                                                false, false);
}

// CDNA5 async DMA: LDS[lds_addr + ...] = MEM[gaddr .. gaddr+15] per lane,
// tracked with ASYNCcnt (CDNA5 ISA).
static __device__ __forceinline__ void async_load_b128(unsigned lds_addr,
                                                       const void* gaddr) {
  asm volatile("global_load_async_to_lds_b128 %0, %1, off"
               :: "v"(lds_addr),
                  "v"((unsigned long long)(uintptr_t)gaddr)
               : "memory");
}

template <int N>
static __device__ __forceinline__ void wait_asynccnt() {
  asm volatile("s_wait_asynccnt %0" :: "n"(N) : "memory");
}

// ---------------------------------------------------------------------------
// 1) Relative-position bias table: btab[h][(j-i)+S-1] = rel_embed[bucket(j-i)][h]
// ---------------------------------------------------------------------------
__global__ void btab_kernel(const float* __restrict__ rel_embed,
                            float* __restrict__ btab) {
  int idx = blockIdx.x * 256 + threadIdx.x;
  const int W = 2 * S_ - 1;
  if (idx >= NH_ * W) return;
  int h = idx / W;
  int r = idx % W;
  int rel = r - (S_ - 1);
  const int nb = NB_ / 2;          // 160
  const int max_exact = nb / 2;    // 80
  int bucket = (rel > 0) ? nb : 0;
  int ar = rel < 0 ? -rel : rel;
  if (ar < max_exact) {
    bucket += ar;
  } else {
    int large = max_exact +
        (int)(logf((float)ar / (float)max_exact) /
              logf((float)MAXD_ / (float)max_exact) * (float)(nb - max_exact));
    large = large < nb - 1 ? large : nb - 1;
    bucket += large;
  }
  btab[h * W + r] = rel_embed[bucket * NH_ + h];
}

// ---------------------------------------------------------------------------
// 2) Gate: g[b,h,s] = sigmoid(z0) * (sigmoid(z1)*gru_const[h] - 1) + 2
// ---------------------------------------------------------------------------
__global__ void gate_kernel(const float* __restrict__ x,
                            const float* __restrict__ Wg,
                            const float* __restrict__ bg,
                            const float* __restrict__ gru_const,
                            float* __restrict__ gate) {
  int idx = blockIdx.x * 256 + threadIdx.x;     // b*NH*S threads
  int b = idx / (NH_ * S_);
  int rem = idx % (NH_ * S_);
  int h = rem / S_;
  int s = rem % S_;
  const float* xp = x + ((size_t)(b * S_ + s)) * H_ + h * HD_;
  float acc[8];
  #pragma unroll
  for (int e = 0; e < 8; ++e) acc[e] = bg[e];
  for (int d = 0; d < HD_; ++d) {
    float xv = xp[d];
    #pragma unroll
    for (int e = 0; e < 8; ++e) acc[e] += xv * Wg[e * HD_ + d];
  }
  float z0 = acc[0] + acc[1] + acc[2] + acc[3];
  float z1 = acc[4] + acc[5] + acc[6] + acc[7];
  float ga = 1.0f / (1.0f + __expf(-z0));
  float gb = 1.0f / (1.0f + __expf(-z1));
  gate[idx] = ga * (gb * gru_const[h] - 1.0f) + 2.0f;
}

// ---------------------------------------------------------------------------
// 3) Generic GEMM: C[M,N] = act(A[M,K] * W[N,K]^T + bias[N])
//    BM=128, BN=64, BK=32; 8 waves, each computes 32x32 via 2x2 WMMA tiles.
//    act: 0=none, 1=GELU(erf), 2=ELU, 3=scale by 0.125 (Q).
//    outHalf: store _Float16 instead of float.
// ---------------------------------------------------------------------------
__global__ __launch_bounds__(256) void gemm_bias_act(
    const float* __restrict__ A, const float* __restrict__ W,
    const float* __restrict__ bias, void* __restrict__ C,
    int M, int N, int K, int act, int outHalf) {
  __shared__ _Float16 As[128][40];
  __shared__ _Float16 Ws[64][40];

  int tid  = threadIdx.x;
  int w    = tid >> 5;
  int lane = tid & 31;
  int half = lane >> 4;
  int nlo  = lane & 15;
  int wm = (w >> 1) * 32;          // 4 wave-rows of 32
  int wn = (w & 1) * 32;           // 2 wave-cols of 32
  int m0 = blockIdx.y * 128;
  int n0 = blockIdx.x * 64;

  v8f zero = {0.f,0.f,0.f,0.f,0.f,0.f,0.f,0.f};
  v8f acc[2][2];
  #pragma unroll
  for (int i = 0; i < 2; ++i)
    #pragma unroll
    for (int j = 0; j < 2; ++j) acc[i][j] = zero;

  int ar = tid >> 1;               // A: 2 threads/row, 16 cols each
  int ac = (tid & 1) * 16;
  int wr = tid >> 2;               // W: 4 threads/row, 8 cols each
  int wc = (tid & 3) * 8;

  for (int k0 = 0; k0 < K; k0 += 32) {
    // prefetch next tiles into the WGP cache while this one computes
    // (locality 3 -> WGP-scope prefetch: pulls into all cache levels)
    if (k0 + 32 < K) {
      __builtin_prefetch(A + (size_t)(m0 + ar) * K + k0 + 32 + ac, 0, 3);
      __builtin_prefetch(W + (size_t)(n0 + wr) * K + k0 + 32 + wc, 0, 3);
    }
    __syncthreads();
    {
      const float4* ap =
          reinterpret_cast<const float4*>(A + (size_t)(m0 + ar) * K + k0 + ac);
      #pragma unroll
      for (int i = 0; i < 4; ++i) {
        float4 f = ap[i];
        As[ar][ac + i * 4 + 0] = (_Float16)f.x;
        As[ar][ac + i * 4 + 1] = (_Float16)f.y;
        As[ar][ac + i * 4 + 2] = (_Float16)f.z;
        As[ar][ac + i * 4 + 3] = (_Float16)f.w;
      }
      const float4* wp =
          reinterpret_cast<const float4*>(W + (size_t)(n0 + wr) * K + k0 + wc);
      #pragma unroll
      for (int i = 0; i < 2; ++i) {
        float4 f = wp[i];
        Ws[wr][wc + i * 4 + 0] = (_Float16)f.x;
        Ws[wr][wc + i * 4 + 1] = (_Float16)f.y;
        Ws[wr][wc + i * 4 + 2] = (_Float16)f.z;
        Ws[wr][wc + i * 4 + 3] = (_Float16)f.w;
      }
    }
    __syncthreads();

    v16h af[2], bf[2];
    #pragma unroll
    for (int mt = 0; mt < 2; ++mt) {
      int mrow = wm + mt * 16 + nlo;
      #pragma unroll
      for (int vv = 0; vv < 8; ++vv) {
        int kk = ((vv >> 2) << 4) + (half << 3) + ((vv & 3) << 1);
        af[mt][2 * vv]     = As[mrow][kk];
        af[mt][2 * vv + 1] = As[mrow][kk + 1];
      }
    }
    #pragma unroll
    for (int nt = 0; nt < 2; ++nt) {
      int nrow = wn + nt * 16 + nlo;
      #pragma unroll
      for (int vv = 0; vv < 8; ++vv) {
        int kk = ((vv >> 2) << 4) + (half << 3) + ((vv & 3) << 1);
        bf[nt][2 * vv]     = Ws[nrow][kk];
        bf[nt][2 * vv + 1] = Ws[nrow][kk + 1];
      }
    }
    #pragma unroll
    for (int mt = 0; mt < 2; ++mt)
      #pragma unroll
      for (int nt = 0; nt < 2; ++nt)
        acc[mt][nt] = wmma_f16(af[mt], bf[nt], acc[mt][nt]);
  }

  #pragma unroll
  for (int mt = 0; mt < 2; ++mt)
    #pragma unroll
    for (int nt = 0; nt < 2; ++nt) {
      int col = n0 + wn + nt * 16 + nlo;
      float bval = bias[col];
      #pragma unroll
      for (int e = 0; e < 8; ++e) {
        int row = m0 + wm + mt * 16 + e + 8 * half;
        float v = acc[mt][nt][e] + bval;
        if (act == 1)      v = 0.5f * v * (1.0f + erff(v * 0.70710678118654752f));
        else if (act == 2) v = v > 0.f ? v : (__expf(v) - 1.0f);
        else if (act == 3) v = v * 0.125f;   // fold 1/sqrt(HD) into Q
        size_t off = (size_t)row * N + col;
        if (outHalf) ((_Float16*)C)[off] = (_Float16)v;
        else         ((float*)C)[off] = v;
      }
    }
}

// ---------------------------------------------------------------------------
// 4) Flash attention: one block = 128 query rows of one (b,h).
//    q,k,v are f16 (q pre-scaled by 0.125). K/V tiles staged into LDS via
//    CDNA5 async DMA (global_load_async_to_lds_b128), double-buffered.
//    scores = q.k + g[i]*btab[h][j-i]; online softmax; o = P.V
// ---------------------------------------------------------------------------
__global__ __launch_bounds__(256) void attn_kernel(
    const _Float16* __restrict__ q, const _Float16* __restrict__ kg,
    const _Float16* __restrict__ vg, const float* __restrict__ gate,
    const float* __restrict__ btab, float* __restrict__ ctx) {
  // row stride 80 halves = 160B (16B aligned rows for B128 async copies)
  __shared__ __align__(16) _Float16 Ks[2][64][80];
  __shared__ __align__(16) _Float16 Vs[2][64][80];
  __shared__ _Float16 Pls[8][16][72];

  int bh = blockIdx.y;
  int b  = bh / NH_;
  int h  = bh % NH_;
  int i0 = blockIdx.x * 128;

  int tid  = threadIdx.x;
  int w    = tid >> 5;
  int lane = tid & 31;
  int half = lane >> 4;
  int nlo  = lane & 15;
  int rowBase = i0 + w * 16;

  // issue async copies of a 64x64 f16 K/V tile pair into buffer `bi`.
  // 64 rows * 128B = 512 chunks of 16B; 8 waves * 2 instr * 32 lanes = 512.
  auto issue_tile = [&](int j0, int bi) {
    #pragma unroll
    for (int t = 0; t < 2; ++t) {
      int chunk = (w * 2 + t) * 32 + lane;       // 0..511
      int row = chunk >> 3;
      int co  = (chunk & 7) * 16;                // byte offset within 128B row
      const char* gk = (const char*)(kg + ((size_t)(b * S_ + j0 + row)) * H_ + h * HD_) + co;
      const char* gv = (const char*)(vg + ((size_t)(b * S_ + j0 + row)) * H_ + h * HD_) + co;
      unsigned lk = (unsigned)(uintptr_t)(&Ks[bi][row][0]) + (unsigned)co;
      unsigned lv = (unsigned)(uintptr_t)(&Vs[bi][row][0]) + (unsigned)co;
      async_load_b128(lk, gk);
      async_load_b128(lv, gv);
    }
  };

  // Q fragments (f16, already scaled by 1/sqrt(64))
  v16h qa0, qa1;
  {
    int i = rowBase + nlo;
    const _Float16* qp = q + ((size_t)(b * S_ + i)) * H_ + h * HD_;
    #pragma unroll
    for (int vv = 0; vv < 8; ++vv) {
      int kk = ((vv >> 2) << 4) + (half << 3) + ((vv & 3) << 1);
      qa0[2 * vv]     = qp[kk];
      qa0[2 * vv + 1] = qp[kk + 1];
      qa1[2 * vv]     = qp[kk + 32];
      qa1[2 * vv + 1] = qp[kk + 33];
    }
  }
  float gv8[8];
  #pragma unroll
  for (int e = 0; e < 8; ++e)
    gv8[e] = gate[(size_t)bh * S_ + rowBase + e + 8 * half];

  v8f zero = {0.f,0.f,0.f,0.f,0.f,0.f,0.f,0.f};
  v8f o[4];
  #pragma unroll
  for (int nt = 0; nt < 4; ++nt) o[nt] = zero;
  float mrow[8], lrow[8];
  #pragma unroll
  for (int e = 0; e < 8; ++e) { mrow[e] = -3.0e38f; lrow[e] = 0.f; }

  const int BW = 2 * S_ - 1;
  const int NBLK = S_ / 64;

  issue_tile(0, 0);                              // prologue: tile 0 -> buf 0

  for (int kb = 0; kb < NBLK; ++kb) {
    int cur = kb & 1;
    __syncthreads();                             // buf cur^1 free to overwrite
    if (kb + 1 < NBLK) {
      issue_tile((kb + 1) * 64, cur ^ 1);        // prefetch next tile (DMA)
      wait_asynccnt<4>();                        // current tile's 4 copies done
    } else {
      wait_asynccnt<0>();
    }
    __syncthreads();                             // all waves' chunks landed

    // scores for 4 column tiles
    v8f st[4];
    #pragma unroll
    for (int nt = 0; nt < 4; ++nt) {
      v16h kb0, kb1;
      int jrow = nt * 16 + nlo;
      #pragma unroll
      for (int vv = 0; vv < 8; ++vv) {
        int kk = ((vv >> 2) << 4) + (half << 3) + ((vv & 3) << 1);
        kb0[2 * vv]     = Ks[cur][jrow][kk];
        kb0[2 * vv + 1] = Ks[cur][jrow][kk + 1];
        kb1[2 * vv]     = Ks[cur][jrow][kk + 32];
        kb1[2 * vv + 1] = Ks[cur][jrow][kk + 33];
      }
      v8f c = zero;
      c = wmma_f16(qa0, kb0, c);
      c = wmma_f16(qa1, kb1, c);
      int j0 = kb * 64;
      #pragma unroll
      for (int e = 0; e < 8; ++e) {
        int i = rowBase + e + 8 * half;
        int j = j0 + nt * 16 + nlo;
        c[e] += gv8[e] * btab[h * BW + (j - i) + (S_ - 1)];
      }
      st[nt] = c;
    }

    // online softmax: row max over this 64-key block
    float mx[8];
    #pragma unroll
    for (int e = 0; e < 8; ++e)
      mx[e] = fmaxf(fmaxf(st[0][e], st[1][e]), fmaxf(st[2][e], st[3][e]));
    #pragma unroll
    for (int m = 1; m <= 8; m <<= 1)
      #pragma unroll
      for (int e = 0; e < 8; ++e)
        mx[e] = fmaxf(mx[e], __shfl_xor(mx[e], m, 32));

    #pragma unroll
    for (int e = 0; e < 8; ++e) {
      float mn = fmaxf(mrow[e], mx[e]);
      float sc = __expf(mrow[e] - mn);
      mrow[e] = mn;
      lrow[e] *= sc;
      o[0][e] *= sc; o[1][e] *= sc; o[2][e] *= sc; o[3][e] *= sc;
    }
    #pragma unroll
    for (int nt = 0; nt < 4; ++nt)
      #pragma unroll
      for (int e = 0; e < 8; ++e) {
        float p = __expf(st[nt][e] - mrow[e]);
        lrow[e] += p;
        Pls[w][e + 8 * half][nt * 16 + nlo] = (_Float16)p;
      }

    // P (16x64) reloaded in A-fragment layout, then o += P.V
    v16h pa0, pa1;
    #pragma unroll
    for (int vv = 0; vv < 8; ++vv) {
      int kk = ((vv >> 2) << 4) + (half << 3) + ((vv & 3) << 1);
      pa0[2 * vv]     = Pls[w][nlo][kk];
      pa0[2 * vv + 1] = Pls[w][nlo][kk + 1];
      pa1[2 * vv]     = Pls[w][nlo][kk + 32];
      pa1[2 * vv + 1] = Pls[w][nlo][kk + 33];
    }
    #pragma unroll
    for (int nt = 0; nt < 4; ++nt) {
      v16h vb0, vb1;
      int d = nt * 16 + nlo;
      #pragma unroll
      for (int vv = 0; vv < 8; ++vv) {
        int kk = ((vv >> 2) << 4) + (half << 3) + ((vv & 3) << 1);
        vb0[2 * vv]     = Vs[cur][kk][d];
        vb0[2 * vv + 1] = Vs[cur][kk + 1][d];
        vb1[2 * vv]     = Vs[cur][kk + 32][d];
        vb1[2 * vv + 1] = Vs[cur][kk + 33][d];
      }
      o[nt] = wmma_f16(pa0, vb0, o[nt]);
      o[nt] = wmma_f16(pa1, vb1, o[nt]);
    }
  }

  // final row sums and normalized store
  #pragma unroll
  for (int m = 1; m <= 8; m <<= 1)
    #pragma unroll
    for (int e = 0; e < 8; ++e)
      lrow[e] += __shfl_xor(lrow[e], m, 32);
  #pragma unroll
  for (int e = 0; e < 8; ++e) lrow[e] = 1.0f / lrow[e];

  #pragma unroll
  for (int nt = 0; nt < 4; ++nt)
    #pragma unroll
    for (int e = 0; e < 8; ++e) {
      int i = rowBase + e + 8 * half;
      int d = nt * 16 + nlo;
      ctx[((size_t)(b * S_ + i)) * H_ + h * HD_ + d] = o[nt][e] * lrow[e];
    }
}

// ---------------------------------------------------------------------------
// 5) Fused add + LayerNorm: out = LN(a + b [+ c]) * g + beta, row = 1024
// ---------------------------------------------------------------------------
__global__ __launch_bounds__(256) void ln3_kernel(
    const float* __restrict__ a, const float* __restrict__ b,
    const float* __restrict__ c, const float* __restrict__ g,
    const float* __restrict__ beta, float* __restrict__ out) {
  int row = blockIdx.x;
  size_t off = (size_t)row * H_;
  int tid = threadIdx.x;
  float vals[4];
  float s1 = 0.f, s2 = 0.f;
  #pragma unroll
  for (int i = 0; i < 4; ++i) {
    int idx = tid + i * 256;
    float vv = a[off + idx] + b[off + idx];
    if (c) vv += c[off + idx];
    vals[i] = vv;
    s1 += vv;
    s2 += vv * vv;
  }
  __shared__ float r1[256], r2[256];
  r1[tid] = s1; r2[tid] = s2;
  __syncthreads();
  for (int st = 128; st > 0; st >>= 1) {
    if (tid < st) { r1[tid] += r1[tid + st]; r2[tid] += r2[tid + st]; }
    __syncthreads();
  }
  float mean = r1[0] * (1.0f / H_);
  float var  = r2[0] * (1.0f / H_) - mean * mean;
  float inv  = rsqrtf(var + EPS_);
  #pragma unroll
  for (int i = 0; i < 4; ++i) {
    int idx = tid + i * 256;
    out[off + idx] = (vals[i] - mean) * inv * g[idx] + beta[idx];
  }
}

// ---------------------------------------------------------------------------
extern "C" void kernel_launch(void* const* d_in, const int* in_sizes, int n_in,
                              void* d_out, int out_size, void* d_ws, size_t ws_size,
                              hipStream_t stream) {
  const float* x    = (const float*)d_in[0];
  const float* Wq   = (const float*)d_in[1];
  const float* bq   = (const float*)d_in[2];
  const float* Wk   = (const float*)d_in[3];
  const float* bk   = (const float*)d_in[4];
  const float* Wv   = (const float*)d_in[5];
  const float* bv   = (const float*)d_in[6];
  const float* Wo   = (const float*)d_in[7];
  const float* bo   = (const float*)d_in[8];
  const float* Wg   = (const float*)d_in[9];
  const float* bg   = (const float*)d_in[10];
  const float* gru  = (const float*)d_in[11];
  const float* rel  = (const float*)d_in[12];
  const float* l1g  = (const float*)d_in[13];
  const float* l1b  = (const float*)d_in[14];
  const float* W1   = (const float*)d_in[15];
  const float* b1   = (const float*)d_in[16];
  const float* W2   = (const float*)d_in[17];
  const float* b2   = (const float*)d_in[18];
  const float* l2g  = (const float*)d_in[19];
  const float* l2b  = (const float*)d_in[20];
  const float* Wad  = (const float*)d_in[21];
  const float* bad  = (const float*)d_in[22];
  const float* Wau  = (const float*)d_in[23];
  const float* bau  = (const float*)d_in[24];

  const size_t M4 = (size_t)B_ * S_ * H_;           // 4M elements
  _Float16* qh = (_Float16*)d_ws;
  _Float16* kh = qh + M4;
  _Float16* vh = kh + M4;
  float* ctx  = (float*)(vh + M4);
  float* h1   = ctx  + M4;
  float* ffm  = h1   + M4;                          // 16M floats
  float* tb   = ffm  + (size_t)B_ * S_ * FF_;
  float* adm  = tb   + M4;                          // 1M floats
  float* gate = adm  + (size_t)B_ * S_ * AD_;
  float* btab = gate + (size_t)B_ * NH_ * S_;

  const int M = B_ * S_;   // 4096

  btab_kernel<<<(NH_ * (2 * S_ - 1) + 255) / 256, 256, 0, stream>>>(rel, btab);
  gate_kernel<<<(B_ * NH_ * S_) / 256, 256, 0, stream>>>(x, Wg, bg, gru, gate);

  // Q, K, V projections -> f16 (Q pre-scaled by 1/sqrt(HD))
  gemm_bias_act<<<dim3(H_ / 64, M / 128), 256, 0, stream>>>(x, Wq, bq, qh, M, H_, H_, 3, 1);
  gemm_bias_act<<<dim3(H_ / 64, M / 128), 256, 0, stream>>>(x, Wk, bk, kh, M, H_, H_, 0, 1);
  gemm_bias_act<<<dim3(H_ / 64, M / 128), 256, 0, stream>>>(x, Wv, bv, vh, M, H_, H_, 0, 1);

  // attention (async-DMA staged K/V, double-buffered)
  attn_kernel<<<dim3(S_ / 128, B_ * NH_), 256, 0, stream>>>(qh, kh, vh, gate, btab, ctx);

  // output projection, residual + LN1
  gemm_bias_act<<<dim3(H_ / 64, M / 128), 256, 0, stream>>>(ctx, Wo, bo, tb, M, H_, H_, 0, 0);
  ln3_kernel<<<M, 256, 0, stream>>>(x, tb, nullptr, l1g, l1b, h1);

  // FFN (GELU) and adapter (ELU)
  gemm_bias_act<<<dim3(FF_ / 64, M / 128), 256, 0, stream>>>(h1, W1, b1, ffm, M, FF_, H_, 1, 0);
  gemm_bias_act<<<dim3(H_ / 64,  M / 128), 256, 0, stream>>>(ffm, W2, b2, tb,  M, H_, FF_, 0, 0);
  gemm_bias_act<<<dim3(AD_ / 64, M / 128), 256, 0, stream>>>(h1, Wad, bad, adm, M, AD_, H_, 2, 0);
  gemm_bias_act<<<dim3(H_ / 64,  M / 128), 256, 0, stream>>>(adm, Wau, bau, ctx, M, H_, AD_, 0, 0);

  // h + ff + adapt, LN2 -> out
  ln3_kernel<<<M, 256, 0, stream>>>(h1, tb, ctx, l2g, l2b, (float*)d_out);
}
